// BahdanauAttention_64415919505755
// MI455X (gfx1250) — matrix operand order repaired and checked by reading, not compiled, not run on hardware
//
#include <hip/hip_runtime.h>
#include <stdint.h>

#define B_  32
#define T_  2048
#define D_  1024
#define U_  1024

typedef __attribute__((ext_vector_type(16))) __bf16 bf16x16;
typedef __attribute__((ext_vector_type(8)))  float  floatx8;

union FragBF { bf16x16 v; uint4 q[2]; };

// A-tile in LDS: 32 rows x 1024 cols bf16, padded to 1032 to spread banks
#define TM      32
#define LDA     1032
#define A_BYTES (TM * LDA * 2)   // 66048

// Branch-free tanh: tanh(x) = sign(x) * (1 - e^{-2|x|}) / (1 + e^{-2|x|})
// __expf -> v_exp_f32, __builtin_amdgcn_rcpf -> v_rcp_f32. No EXEC divergence.
__device__ __forceinline__ float fast_tanh(float x) {
    const float ax = __builtin_fabsf(x);
    const float t  = __expf(-2.0f * ax);
    const float r  = (1.0f - t) * __builtin_amdgcn_rcpf(1.0f + t);
    return __builtin_copysignf(r, x);
}

// ---------------------------------------------------------------------------
// Kernel 1: W2 [D,U] fp32 -> W2^T [U,D] bf16 (LDS tiled transpose)
// ---------------------------------------------------------------------------
__global__ __launch_bounds__(256) void w2t_kernel(const float* __restrict__ W2,
                                                  __bf16* __restrict__ w2t) {
    __shared__ float tile[32][33];
    const int u0 = blockIdx.x * 32;
    const int d0 = blockIdx.y * 32;
    const int tx = threadIdx.x & 31;
    const int ty = threadIdx.x >> 5;      // 8 rows of the tile at a time
#pragma unroll
    for (int j = 0; j < 32; j += 8)
        tile[ty + j][tx] = W2[(size_t)(d0 + ty + j) * U_ + (u0 + tx)];
    __syncthreads();
#pragma unroll
    for (int j = 0; j < 32; j += 8)
        w2t[(size_t)(u0 + ty + j) * D_ + (d0 + tx)] = (__bf16)tile[tx][ty + j];
}

// ---------------------------------------------------------------------------
// Kernel 2: q_proj[b][u] = sum_d query[b][d] * W1[d][u]   (fp32, tiny)
// ---------------------------------------------------------------------------
__global__ __launch_bounds__(256) void qproj_kernel(const float* __restrict__ query,
                                                    const float* __restrict__ W1,
                                                    float* __restrict__ qproj) {
    const int b = blockIdx.y;
    const int u = blockIdx.x * 256 + threadIdx.x;
    __shared__ float q[D_];
    for (int i = threadIdx.x; i < D_; i += 256) q[i] = query[(size_t)b * D_ + i];
    __syncthreads();
    float acc = 0.0f;
#pragma unroll 4
    for (int d = 0; d < D_; ++d) acc += q[d] * W1[(size_t)d * U_ + u];
    qproj[(size_t)b * U_ + u] = acc;
}

// ---------------------------------------------------------------------------
// Kernel 3: logits[b][t] = sum_u tanh(qproj[b][u] + (values @ W2)[b][t][u]) * V[u]
// WMMA bf16: each WG = 8 wave32s in a 2(M) x 4(N) arrangement over a 32x64 tile.
// ---------------------------------------------------------------------------
__global__ __launch_bounds__(256) void attn_logits_kernel(
    const float*  __restrict__ values,
    const __bf16* __restrict__ w2t,     // [U][D] bf16
    const float*  __restrict__ qproj,   // [B][U]
    const float*  __restrict__ V,       // [U]
    float*        __restrict__ logits)  // [B][T]
{
    extern __shared__ char smem[];
    __bf16* As  = (__bf16*)smem;                       // [TM][LDA] bf16
    float*  qv  = (float*)(smem + A_BYTES);            // 1024
    float*  vvs = qv + U_;                             // 1024
    float*  lt  = vvs + U_;                            // 32

    const int tid = threadIdx.x;
    const int b   = blockIdx.y;
    const int t0  = blockIdx.x * TM;

    for (int i = tid; i < U_; i += 256) {
        qv[i]  = qproj[(size_t)b * U_ + i];
        vvs[i] = V[i];
    }
    if (tid < TM) lt[tid] = 0.0f;

    // Load values tile [TM x D] fp32 -> bf16 LDS (float4 reads, 8B LDS stores)
    const float4* src = (const float4*)(values + (size_t)b * T_ * D_ + (size_t)t0 * D_);
    for (int idx = tid; idx < TM * (D_ / 4); idx += 256) {
        const int row = idx >> 8;          // D_/4 == 256 float4 per row
        const int c4  = idx & 255;
        float4 f = src[(size_t)row * (D_ / 4) + c4];
        union { __bf16 h[4]; uint2 u2; } p;
        p.h[0] = (__bf16)f.x; p.h[1] = (__bf16)f.y;
        p.h[2] = (__bf16)f.z; p.h[3] = (__bf16)f.w;
        *(uint2*)&As[row * LDA + c4 * 4] = p.u2;
    }
    __syncthreads();

    const int lane  = tid & 31;
    const int wave  = tid >> 5;
    const int waveM = wave & 1;            // 0..1 -> rows m0 = waveM*16
    const int waveN = wave >> 1;           // 0..3 -> cols n0 = waveN*16
    const int ln    = lane & 15;
    const int lh    = lane >> 4;           // half-wave select
    const int arow  = waveM * 16 + ln;     // A row this lane supplies

    float partial[8];
#pragma unroll
    for (int i = 0; i < 8; ++i) partial[i] = 0.0f;

    for (int nt = 0; nt < U_ / 64; ++nt) {
        const int ncol = nt * 64 + waveN * 16 + ln;       // B column this lane supplies
        const __bf16* bptr = w2t + (size_t)ncol * D_ + lh * 8;
        floatx8 acc = {0.f, 0.f, 0.f, 0.f, 0.f, 0.f, 0.f, 0.f};
#pragma unroll 4
        for (int k = 0; k < D_; k += 32) {
            FragBF a, bb;
            // A 16x32 bf16: lane = row m, chunks K = k + lh*8 .. +7 and k+16+lh*8 .. +7
            a.q[0] = *(const uint4*)&As[arow * LDA + k + lh * 8];
            a.q[1] = *(const uint4*)&As[arow * LDA + k + 16 + lh * 8];
            // B 32x16 bf16: lane = col n, chunks K = k + lh*8 .. +7 and k+16+lh*8 .. +7
            const uint4* bp = (const uint4*)(bptr + k);
            bb.q[0] = bp[0];
            bb.q[1] = bp[2];
            acc = __builtin_amdgcn_wmma_f32_16x16x32_bf16(
                false, a.v, false, bb.v, (short)0, acc, false, false);
        }
        // C layout: element i of acc -> (m = waveM*16 + i + lh*8, n = ncol)
        const float qb = qv[ncol];
        const float vb = vvs[ncol];
#pragma unroll
        for (int i = 0; i < 8; ++i)
            partial[i] += fast_tanh(acc[i] + qb) * vb;
    }

    // Reduce each row-partial across the 16 lanes of each half-wave
#pragma unroll
    for (int i = 0; i < 8; ++i) {
        float p = partial[i];
        p += __shfl_xor(p, 1);
        p += __shfl_xor(p, 2);
        p += __shfl_xor(p, 4);
        p += __shfl_xor(p, 8);
        partial[i] = p;
    }
    if (ln == 0) {
        const int mb = waveM * 16 + lh * 8;
#pragma unroll
        for (int i = 0; i < 8; ++i) atomicAdd(&lt[mb + i], partial[i]);
    }
    __syncthreads();
    if (tid < TM) logits[(size_t)b * T_ + t0 + tid] = lt[tid];
}

// ---------------------------------------------------------------------------
// Kernel 4: softmax over T per batch; writes weights to d_out[32768:],
// zeroes the context region d_out[0:32768] (harness poisons d_out).
// ---------------------------------------------------------------------------
__global__ __launch_bounds__(256) void softmax_kernel(const float* __restrict__ logits,
                                                      float* __restrict__ out) {
    const int b = blockIdx.x;
    const int tid = threadIdx.x;
    __shared__ float e[T_];
    __shared__ float red[256];

    float mx = -3.402823466e38f;
    for (int t = tid; t < T_; t += 256) {
        float l = logits[(size_t)b * T_ + t];
        e[t] = l;
        mx = fmaxf(mx, l);
    }
    red[tid] = mx;
    __syncthreads();
    for (int s = 128; s > 0; s >>= 1) {
        if (tid < s) red[tid] = fmaxf(red[tid], red[tid + s]);
        __syncthreads();
    }
    mx = red[0];
    __syncthreads();

    float sum = 0.0f;
    for (int t = tid; t < T_; t += 256) {
        float ex = expf(e[t] - mx);
        e[t] = ex;
        sum += ex;
    }
    red[tid] = sum;
    __syncthreads();
    for (int s = 128; s > 0; s >>= 1) {
        if (tid < s) red[tid] += red[tid + s];
        __syncthreads();
    }
    const float inv = 1.0f / red[0];

    for (int t = tid; t < T_; t += 256)
        out[B_ * D_ + (size_t)b * T_ + t] = e[t] * inv;
    // zero context row b for the atomic accumulation in kernel 5
    for (int i = tid; i < D_; i += 256)
        out[(size_t)b * D_ + i] = 0.0f;
}

// ---------------------------------------------------------------------------
// Kernel 5: context[b][d] = sum_t w[b][t] * values[b][t][d]  (T split 8-way)
// ---------------------------------------------------------------------------
__global__ __launch_bounds__(256) void context_kernel(const float* __restrict__ values,
                                                      float* __restrict__ out) {
    const int tid = threadIdx.x;
    const int d   = blockIdx.x * 256 + tid;
    const int b   = blockIdx.y;
    const int t0  = blockIdx.z * 256;
    __shared__ float w[256];
    w[tid] = out[B_ * D_ + (size_t)b * T_ + t0 + tid];
    __syncthreads();
    const float* vp = values + (size_t)b * T_ * D_ + (size_t)t0 * D_ + d;
    float acc = 0.0f;
#pragma unroll 4
    for (int t = 0; t < 256; ++t) acc += w[t] * vp[(size_t)t * D_];
    atomicAdd(&out[(size_t)b * D_ + d], acc);
}

// ---------------------------------------------------------------------------
extern "C" void kernel_launch(void* const* d_in, const int* in_sizes, int n_in,
                              void* d_out, int out_size, void* d_ws, size_t ws_size,
                              hipStream_t stream) {
    const float* query  = (const float*)d_in[0];   // [B, D]
    const float* values = (const float*)d_in[1];   // [B, T, D]
    const float* W1     = (const float*)d_in[2];   // [D, U]
    const float* W2     = (const float*)d_in[3];   // [D, U]
    const float* V      = (const float*)d_in[4];   // [U, 1]
    float* out = (float*)d_out;                    // [B*D context | B*T weights]

    char* ws = (char*)d_ws;
    __bf16* w2t   = (__bf16*)ws;                           // 2 MB
    float*  qproj = (float*)(ws + (size_t)U_ * D_ * 2);    // 128 KB
    float*  logit = (float*)(ws + (size_t)U_ * D_ * 2 + (size_t)B_ * U_ * 4); // 256 KB

    w2t_kernel<<<dim3(U_ / 32, D_ / 32), 256, 0, stream>>>(W2, w2t);
    qproj_kernel<<<dim3(U_ / 256, B_), 256, 0, stream>>>(query, W1, qproj);

    const size_t smem = A_BYTES + U_ * 4 + U_ * 4 + TM * 4;   // ~74.4 KB
    attn_logits_kernel<<<dim3(T_ / TM, B_), 256, smem, stream>>>(
        values, w2t, qproj, V, logit);

    softmax_kernel<<<B_, 256, 0, stream>>>(logit, out);
    context_kernel<<<dim3(D_ / 256, B_, T_ / 256), 256, 0, stream>>>(values, out);
}
